// GNN_23742579212622
// MI455X (gfx1250) — compile-verified, matrix operand order checked by reading, not run on hardware
//
#include <hip/hip_runtime.h>
#include <hip/hip_bf16.h>
#include <math.h>

typedef __attribute__((ext_vector_type(16))) _Float16 v16h;
typedef __attribute__((ext_vector_type(8)))  float    v8f;

// ---------------- degree / normalization ----------------

__global__ void k_deg_init(float* deg, int n) {
  int i = blockIdx.x * blockDim.x + threadIdx.x;
  if (i < n) deg[i] = 1.0f;                 // self-loop contributes 1
}

__global__ void k_deg_acc(const int* __restrict__ dst, float* deg, int e) {
  int i = blockIdx.x * blockDim.x + threadIdx.x;
  if (i < e) atomicAdd(&deg[dst[i]], 1.0f);
}

__global__ void k_dinv(float* deg, int n) {
  int i = blockIdx.x * blockDim.x + threadIdx.x;
  if (i < n) deg[i] = rsqrtf(deg[i]);       // deg >= 1 always
}

// ---------------- pack W (f32 KxN row-major) -> f16 WMMA B layout ----------------
// packed index = ((nt*numKt + kt)*32 + lane)*16 + j
// lane<16 : col = nt*16+lane,     j<8 -> k = kt*32 + j,      j>=8 -> k = kt*32 + 16 + (j-8)
// lane>=16: col = nt*16+lane-16,  j<8 -> k = kt*32 + 8 + j,  j>=8 -> k = kt*32 + 24 + (j-8)

__global__ void k_pack_w(const float* __restrict__ W, _Float16* __restrict__ out,
                         int K, int Ncols) {
  int tid = blockIdx.x * blockDim.x + threadIdx.x;
  int total = K * Ncols;
  if (tid >= total) return;
  int j    = tid & 15;
  int lane = (tid >> 4) & 31;
  int t    = tid >> 9;
  int numKt = K >> 5;
  int nt = t / numKt;
  int kt = t - nt * numKt;
  int base = (lane >= 16) ? 8 : 0;
  int k = kt * 32 + base + ((j < 8) ? j : (j + 8));
  int n = nt * 16 + (lane & 15);
  out[tid] = (_Float16)W[k * Ncols + n];
}

// ---------------- register-blocked WMMA GEMM ----------------
// out[Nrows x (NUMNT*16)] = A[Nrows x 128] @ W (+bias)
// One wave computes a full 16-row x NUMNT*16-col panel: A converted once,
// reused across NUMNT WMMAs per k-step (4*NUMNT WMMAs per wave total).
// 256 threads = 8 waves per block -> 128 rows per block.

template <int NUMNT>
__global__ void k_gemm_wmma(const float* __restrict__ A, const _Float16* __restrict__ Bpk,
                            const float* __restrict__ bias, float* __restrict__ out,
                            int Nrows) {
  const int Ncols = NUMNT * 16;
  const int wave  = threadIdx.x >> 5;
  const int lane  = threadIdx.x & 31;
  const int row0  = (blockIdx.x * 8 + wave) * 16;
  if (row0 >= Nrows) return;
  int arow = row0 + (lane & 15);
  if (arow >= Nrows) arow = Nrows - 1;
  const int base = (lane >= 16) ? 8 : 0;
  const float*    rp = A + (size_t)arow * 128 + base;
  const _Float16* bp = Bpk + (size_t)lane * 16;

  v8f c[NUMNT];
#pragma unroll
  for (int nt = 0; nt < NUMNT; ++nt) c[nt] = (v8f){0.f,0.f,0.f,0.f,0.f,0.f,0.f,0.f};

#pragma unroll
  for (int kt = 0; kt < 4; ++kt) {
    const float* p = rp + kt * 32;
    float4 f0 = *(const float4*)(p + 0);
    float4 f1 = *(const float4*)(p + 4);
    float4 f2 = *(const float4*)(p + 16);
    float4 f3 = *(const float4*)(p + 20);
    v16h a;
    a[0]=(_Float16)f0.x;  a[1]=(_Float16)f0.y;  a[2]=(_Float16)f0.z;  a[3]=(_Float16)f0.w;
    a[4]=(_Float16)f1.x;  a[5]=(_Float16)f1.y;  a[6]=(_Float16)f1.z;  a[7]=(_Float16)f1.w;
    a[8]=(_Float16)f2.x;  a[9]=(_Float16)f2.y;  a[10]=(_Float16)f2.z; a[11]=(_Float16)f2.w;
    a[12]=(_Float16)f3.x; a[13]=(_Float16)f3.y; a[14]=(_Float16)f3.z; a[15]=(_Float16)f3.w;
#pragma unroll
    for (int nt = 0; nt < NUMNT; ++nt) {
      v16h b = *(const v16h*)(bp + ((size_t)nt * 4 + kt) * 512);
      c[nt] = __builtin_amdgcn_wmma_f32_16x16x32_f16(false, a, false, b, (short)0,
                                                     c[nt], false, false);
    }
  }

  const int rbase = row0 + ((lane >= 16) ? 8 : 0);
#pragma unroll
  for (int nt = 0; nt < NUMNT; ++nt) {
    const int col = nt * 16 + (lane & 15);
    const float bv = bias ? bias[col] : 0.0f;
#pragma unroll
    for (int r = 0; r < 8; ++r) {
      int rr = rbase + r;
      if (rr < Nrows) out[(size_t)rr * Ncols + col] = c[nt][r] + bv;
    }
  }
}

// ---------------- aggregation: init with bias + self-loop, scatter edges, relu ----------------

__global__ void k_agg_init(const float* __restrict__ t, const float* __restrict__ dinv,
                           const float* __restrict__ bias, float* __restrict__ agg, int Nn) {
  int i = blockIdx.x * blockDim.x + threadIdx.x;          // one float4 group per thread
  if (i >= Nn * 32) return;
  int node = i >> 5, g = i & 31;
  float d = dinv[node];
  float w = d * d;                                        // self-loop norm
  float4 tv = *(const float4*)(t + (size_t)node * 128 + g * 4);
  float4 bv = *(const float4*)(bias + g * 4);
  float4 o;
  o.x = bv.x + w * tv.x;  o.y = bv.y + w * tv.y;
  o.z = bv.z + w * tv.z;  o.w = bv.w + w * tv.w;
  *(float4*)(agg + (size_t)node * 128 + g * 4) = o;
}

__global__ void k_scatter(const int* __restrict__ src, const int* __restrict__ dst,
                          const float* __restrict__ dinv, const float* __restrict__ t,
                          float* __restrict__ agg, int E) {
  int e = blockIdx.x * 8 + (threadIdx.x >> 5);            // one wave per edge
  if (e >= E) return;
  int lane = threadIdx.x & 31;
  int s = src[e], d = dst[e];
  float nrm = dinv[s] * dinv[d];
  float4 v = *(const float4*)(t + (size_t)s * 128 + lane * 4);
  float* ap = agg + (size_t)d * 128 + lane * 4;
  atomicAdd(ap + 0, nrm * v.x);
  atomicAdd(ap + 1, nrm * v.y);
  atomicAdd(ap + 2, nrm * v.z);
  atomicAdd(ap + 3, nrm * v.w);
}

__global__ void k_relu(float* __restrict__ a, int Nn) {
  int i = blockIdx.x * blockDim.x + threadIdx.x;
  if (i >= Nn * 32) return;
  float4 v = *(float4*)(a + (size_t)i * 4);
  v.x = fmaxf(v.x, 0.f); v.y = fmaxf(v.y, 0.f);
  v.z = fmaxf(v.z, 0.f); v.w = fmaxf(v.w, 0.f);
  *(float4*)(a + (size_t)i * 4) = v;
}

// ---------------- log_softmax over rows of 64 (one wave per row) ----------------

__global__ void k_logsoftmax(float* __restrict__ io, int Nn) {
  int row = blockIdx.x * 8 + (threadIdx.x >> 5);
  if (row >= Nn) return;
  int lane = threadIdx.x & 31;
  float* p = io + (size_t)row * 64;
  float v0 = p[lane], v1 = p[lane + 32];
  float m = fmaxf(v0, v1);
#pragma unroll
  for (int off = 16; off > 0; off >>= 1) m = fmaxf(m, __shfl_xor(m, off, 32));
  float s = expf(v0 - m) + expf(v1 - m);
#pragma unroll
  for (int off = 16; off > 0; off >>= 1) s += __shfl_xor(s, off, 32);
  float lse = m + logf(s);
  p[lane]      = v0 - lse;
  p[lane + 32] = v1 - lse;
}

// ---------------- launcher ----------------

static inline int cdiv(int a, int b) { return (a + b - 1) / b; }

extern "C" void kernel_launch(void* const* d_in, const int* in_sizes, int n_in,
                              void* d_out, int out_size, void* d_ws, size_t ws_size,
                              hipStream_t stream) {
  (void)n_in; (void)out_size; (void)ws_size;
  const float* x  = (const float*)d_in[0];
  const int*   ei = (const int*)d_in[1];
  const int Nn = in_sizes[0] / 128;
  const int E  = in_sizes[1] / 2;
  const int* src = ei;
  const int* dst = ei + E;
  const float* W[4]  = {(const float*)d_in[2], (const float*)d_in[4],
                        (const float*)d_in[6], (const float*)d_in[8]};
  const float* bb[4] = {(const float*)d_in[3], (const float*)d_in[5],
                        (const float*)d_in[7], (const float*)d_in[9]};
  const float* Wm1 = (const float*)d_in[10];
  const float* bm1 = (const float*)d_in[11];
  const float* Wm2 = (const float*)d_in[12];
  const float* bm2 = (const float*)d_in[13];
  float* outF = (float*)d_out;

  // workspace layout (floats): dinv | bufA (Nn*128) | bufB (Nn*128) | packed f16 weights
  float* ws   = (float*)d_ws;
  size_t dsz  = ((size_t)Nn + 255) & ~(size_t)255;
  float* dinv = ws;
  size_t feat = (size_t)Nn * 128;
  float* bufA = ws + dsz;
  float* bufB = bufA + feat;
  _Float16* pk = (_Float16*)(bufB + feat);
  _Float16* pkW[4];
  for (int i = 0; i < 4; ++i) pkW[i] = pk + (size_t)i * 16384;
  _Float16* pkM1 = pk + (size_t)4 * 16384;
  _Float16* pkM2 = pk + (size_t)5 * 16384;

  // normalization: deg (with self-loop) -> dinv, in place
  k_deg_init<<<cdiv(Nn, 256), 256, 0, stream>>>(dinv, Nn);
  k_deg_acc<<<cdiv(E, 256), 256, 0, stream>>>(dst, dinv, E);
  k_dinv<<<cdiv(Nn, 256), 256, 0, stream>>>(dinv, Nn);

  // pack weights into WMMA B layout (f16)
  for (int i = 0; i < 4; ++i)
    k_pack_w<<<cdiv(128 * 128, 256), 256, 0, stream>>>(W[i], pkW[i], 128, 128);
  k_pack_w<<<cdiv(128 * 128, 256), 256, 0, stream>>>(Wm1, pkM1, 128, 128);
  k_pack_w<<<cdiv(128 * 64, 256), 256, 0, stream>>>(Wm2, pkM2, 128, 64);

  const int gemmGrid = cdiv(Nn, 128);  // 8 waves/block, 16 rows/wave

  // 4 GCN layers: t = h@W ; agg = bias + dinv^2*t ; agg += scatter(norm * t[src]) ; h = relu(agg)
  for (int l = 0; l < 4; ++l) {
    const float* hin = (l == 0) ? x : bufA;
    k_gemm_wmma<8><<<gemmGrid, 256, 0, stream>>>(hin, pkW[l], nullptr, bufB, Nn);
    k_agg_init<<<cdiv(Nn * 32, 256), 256, 0, stream>>>(bufB, dinv, bb[l], bufA, Nn);
    k_scatter<<<cdiv(E, 8), 256, 0, stream>>>(src, dst, dinv, bufB, bufA, E);
    k_relu<<<cdiv(Nn * 32, 256), 256, 0, stream>>>(bufA, Nn);
  }

  // post-MP MLP + log_softmax
  k_gemm_wmma<8><<<gemmGrid, 256, 0, stream>>>(bufA, pkM1, bm1, bufB, Nn);
  k_gemm_wmma<4><<<gemmGrid, 256, 0, stream>>>(bufB, pkM2, bm2, outF, Nn);
  k_logsoftmax<<<cdiv(Nn, 8), 256, 0, stream>>>(outF, Nn);
}
